// LinearAttention_38147899523283
// MI455X (gfx1250) — matrix operand order repaired and checked by reading, not compiled, not run on hardware
//
#include <hip/hip_runtime.h>

// ---------------------------------------------------------------------------
// LinearAttention (relu + L1-norm attention), fused WMMA implementation.
// B=4, C=256, H=W=64 -> N=4096, CR=32 (== WMMA bf16 K dimension, perfect fit)
// ---------------------------------------------------------------------------

#define B_DIM   4
#define C_DIM   256
#define CR_DIM  32
#define N_PIX   4096
#define MT      64          // output columns per block in the fused kernel
#define NSTEP   32          // n-tile (K of the second GEMM)

typedef __attribute__((ext_vector_type(16))) __bf16 v16bf;
typedef __attribute__((ext_vector_type(8)))  __bf16 v8bf;
typedef __attribute__((ext_vector_type(8)))  float  v8f;

// Build a v16bf operand from two 16-byte chunks (compiler emits 2x b128 loads)
__device__ __forceinline__ v16bf load16(const __bf16* p0, const __bf16* p1) {
    v8bf lo = *(const v8bf*)p0;
    v8bf hi = *(const v8bf*)p1;
    v16bf r;
#pragma unroll
    for (int i = 0; i < 8; ++i) { r[i] = lo[i]; r[i + 8] = hi[i]; }
    return r;
}

__device__ __forceinline__ v8f wmma_bf16(v16bf a, v16bf b, v8f c) {
    return __builtin_amdgcn_wmma_f32_16x16x32_bf16(
        /*neg_a=*/false, a, /*neg_b=*/false, b,
        /*c_mod=*/(short)0, c, /*reuse_a=*/false, /*reuse_b=*/false);
}

__device__ __forceinline__ v8f zero8() {
    v8f z = {0.f, 0.f, 0.f, 0.f, 0.f, 0.f, 0.f, 0.f};
    return z;
}

// ---------------------------------------------------------------------------
// Prep 1: convert weights f32 -> bf16 (row-major layouts preserved)
// ---------------------------------------------------------------------------
__global__ __launch_bounds__(256) void cvt_weights(
    const float* __restrict__ Wq, const float* __restrict__ Wk,
    const float* __restrict__ Wv,
    __bf16* __restrict__ Wqb, __bf16* __restrict__ Wkb,
    __bf16* __restrict__ Wvb)
{
    int i = blockIdx.x * 256 + threadIdx.x;
    if (i < CR_DIM * C_DIM) Wqb[i] = (__bf16)Wq[i];
    if (i < CR_DIM * C_DIM) Wkb[i] = (__bf16)Wk[i];
    if (i < C_DIM * C_DIM)  Wvb[i] = (__bf16)Wv[i];
}

// ---------------------------------------------------------------------------
// Prep 2: x [b][c][n] f32  ->  xt [b][n][c] bf16  (LDS tiled transpose)
// grid: B * (C/64) * (N/64) = 1024 blocks, 256 threads
// ---------------------------------------------------------------------------
__global__ __launch_bounds__(256) void transpose_cvt(
    const float* __restrict__ x, __bf16* __restrict__ xt)
{
    __shared__ float t[64][65];
    int blk = blockIdx.x;
    int b  = blk >> 8;       // 256 tiles per batch
    int r  = blk & 255;
    int cb = r >> 6;         // 4 c-tiles
    int nb = r & 63;         // 64 n-tiles
    const float* xb = x + ((size_t)b * C_DIM + cb * 64) * N_PIX + nb * 64;
#pragma unroll
    for (int i = 0; i < 16; ++i) {
        int idx = threadIdx.x + i * 256;
        int c = idx >> 6, n = idx & 63;
        t[c][n] = xb[(size_t)c * N_PIX + n];
    }
    __syncthreads();
    __bf16* xtb = xt + ((size_t)b * N_PIX + nb * 64) * C_DIM + cb * 64;
#pragma unroll
    for (int i = 0; i < 16; ++i) {
        int idx = threadIdx.x + i * 256;
        int n = idx >> 6, c = idx & 63;
        xtb[(size_t)n * C_DIM + c] = (__bf16)t[c][n];
    }
}

// ---------------------------------------------------------------------------
// Prep 3: generic WMMA NT-GEMM:  G[M][Ncols] = L[M][Kd] * R[Ncols][Kd]^T + bias
// One wave per 16x16 output tile, K-loop in steps of 32 (bf16 WMMA).
// Used for: Qt/Kt = xt * Wq^T  (bias per column)  and  Vb = Wv * x (bias per row)
// ---------------------------------------------------------------------------
__global__ __launch_bounds__(256) void gemm_nt_bias(
    const __bf16* __restrict__ L, const __bf16* __restrict__ R,
    const float* __restrict__ bias, __bf16* __restrict__ G,
    int M, int Ncols, int Kd,
    long sL, long sR, long sG, int biasPerRow)
{
    const int lane = threadIdx.x & 31;
    const int l16  = lane & 15;
    const int hi   = lane >> 4;
    const int b    = blockIdx.y;

    int waveId = blockIdx.x * 8 + (threadIdx.x >> 5);
    int tilesN = Ncols >> 4;
    int tm = waveId / tilesN;
    int tn = waveId - tm * tilesN;
    if (tm >= (M >> 4)) return;

    const __bf16* Lb = L + (size_t)b * sL;
    const __bf16* Rb = R + (size_t)b * sR;
    __bf16*       Gb = G + (size_t)b * sG;

    const int arow = tm * 16 + l16;   // A: lanes 0-15 and 16-31 share row index
    const int bcol = tn * 16 + l16;   // B: lane holds one output column

    v8f acc = zero8();
    for (int k0 = 0; k0 < Kd; k0 += 32) {
        // A operand (16x32): lane row arow; K halves {hi*8..+8} and {hi*8+16..+8}
        const __bf16* lp = Lb + (size_t)arow * Kd + k0 + hi * 8;
        v16bf a = load16(lp, lp + 16);
        // B operand (32x16): lane col bcol; K halves {hi*8..+8} and {hi*8+16..+8}
        const __bf16* rp = Rb + (size_t)bcol * Kd + k0 + hi * 8;
        v16bf bb = load16(rp, rp + 16);
        acc = wmma_bf16(a, bb, acc);
    }

    const float cbias = biasPerRow ? 0.f : bias[bcol];
#pragma unroll
    for (int i = 0; i < 8; ++i) {
        int row = tm * 16 + 8 * hi + i;
        float v = acc[i] + (biasPerRow ? bias[row] : cbias);
        Gb[(size_t)row * Ncols + bcol] = (__bf16)v;
    }
}

// ---------------------------------------------------------------------------
// Fused attention kernel.
// Block = 256 threads (8 waves), handles 64 output columns (m) for one batch.
//   Producer phase (per 32-wide n step): each wave computes one 16x16 tile of
//     S = K^T Q via WMMA (K=32 == CR), relu, per-column sums kept in regs,
//     bf16-converted tile stored to LDS P[m][n].
//   Consumer phase: each wave owns 32 C-rows x 64 m-cols of the output,
//     accumulates O += V_tile * P via 8 WMMAs (A from global Vb, B from LDS P).
//   Epilogue: out = gamma * O / max(colsum, eps) + x.
// ---------------------------------------------------------------------------
__global__ __launch_bounds__(256) void attn_fused(
    const __bf16* __restrict__ Qt,   // [B][N][CR]
    const __bf16* __restrict__ Kt,   // [B][N][CR]
    const __bf16* __restrict__ Vb,   // [B][C][N]
    const float*  __restrict__ x,    // [B][C][N]
    const float*  __restrict__ gamma,
    float*        __restrict__ out)  // [B][C][N]
{
    __shared__ __bf16 P[MT][NSTEP];   // relu'd attention tile, [m][n], 4KB
    __shared__ float  dpart[4][MT];   // denominator partials (deterministic)

    const int lane = threadIdx.x & 31;
    const int wave = threadIdx.x >> 5;
    const int l16  = lane & 15;
    const int hi   = lane >> 4;

    const int b      = blockIdx.y;
    const int m_base = blockIdx.x * MT;

    const __bf16* Qtb = Qt + (size_t)b * N_PIX * CR_DIM;
    const __bf16* Ktb = Kt + (size_t)b * N_PIX * CR_DIM;
    const __bf16* Vbb = Vb + (size_t)b * C_DIM * N_PIX;

    // producer tile assignment: ni in {0,1} (n sub-tile), mi_p in {0..3}
    const int ni   = wave & 1;
    const int mi_p = wave >> 1;
    const int mloc_p = 16 * mi_p + l16;           // local m column this lane owns in S

    // Q B-operand is n-invariant: load once.
    v16bf qb;
    {
        const __bf16* qp = Qtb + (size_t)(m_base + mloc_p) * CR_DIM + hi * 8;
        qb = load16(qp, qp + 16);
    }

    // consumer: this wave owns C rows [c_base, c_base+32)
    const int c_base = wave * 32;

    v8f acc[2][4];
#pragma unroll
    for (int ci = 0; ci < 2; ++ci)
#pragma unroll
        for (int mi = 0; mi < 4; ++mi) acc[ci][mi] = zero8();

    float dsum = 0.f;   // this lane's running contribution to denom[mloc_p]

    for (int nb = 0; nb < N_PIX / NSTEP; ++nb) {
        const int n0 = nb * NSTEP;

        // ---- producer: S tile = (K^T)(16n x 32r) * Q(32r x 16m) ----
        {
            const int nrow = n0 + 16 * ni + l16;
            const __bf16* kp = Ktb + (size_t)nrow * CR_DIM + hi * 8;
            v16bf ka = load16(kp, kp + 16);
            v8f s = wmma_bf16(ka, qb, zero8());

            v8bf pb;
            float cs = 0.f;
#pragma unroll
            for (int i = 0; i < 8; ++i) {
                float v = fmaxf(s[i], 0.f);       // relu
                cs += v;
                pb[i] = (__bf16)v;
            }
            dsum += cs;
            // C/D layout: lane holds column mloc_p, rows 8*hi..8*hi+7 of sub-tile
            *(v8bf*)&P[mloc_p][16 * ni + 8 * hi] = pb;
        }
        __syncthreads();

        // ---- consumer: O(c, m) += V(16c x 32n) * P(32n x 16m) ----
        {
            v16bf va[2];
#pragma unroll
            for (int ci = 0; ci < 2; ++ci) {
                const __bf16* vp =
                    Vbb + (size_t)(c_base + 16 * ci + l16) * N_PIX + n0 + hi * 8;
                va[ci] = load16(vp, vp + 16);
            }
#pragma unroll
            for (int mi = 0; mi < 4; ++mi) {
                const __bf16* pp = &P[16 * mi + l16][hi * 8];
                v16bf pbop = load16(pp, pp + 16);
#pragma unroll
                for (int ci = 0; ci < 2; ++ci)
                    acc[ci][mi] = wmma_bf16(va[ci], pbop, acc[ci][mi]);
            }
        }
        __syncthreads();
    }

    // deterministic 4-way denominator reduction through LDS
    dpart[ni * 2 + hi][mloc_p] = dsum;
    __syncthreads();

    const float g = gamma[0];
#pragma unroll
    for (int mi = 0; mi < 4; ++mi) {
        const int ml = 16 * mi + l16;
        const int m  = m_base + ml;
        const float den = fmaxf(dpart[0][ml] + dpart[1][ml] +
                                dpart[2][ml] + dpart[3][ml], 1e-12f);
        const float sc = g / den;
#pragma unroll
        for (int ci = 0; ci < 2; ++ci) {
            const int c0 = c_base + 16 * ci + 8 * hi;
#pragma unroll
            for (int i = 0; i < 8; ++i) {
                const size_t idx = ((size_t)b * C_DIM + c0 + i) * N_PIX + m;
                out[idx] = sc * acc[ci][mi][i] + x[idx];
            }
        }
    }
}

// ---------------------------------------------------------------------------
// Launcher
// inputs: x, Wq, bq, Wk, bk, Wv, bv, gamma  (all f32)
// ---------------------------------------------------------------------------
extern "C" void kernel_launch(void* const* d_in, const int* in_sizes, int n_in,
                              void* d_out, int out_size, void* d_ws, size_t ws_size,
                              hipStream_t stream) {
    (void)in_sizes; (void)n_in; (void)out_size; (void)ws_size;
    const float* x     = (const float*)d_in[0];
    const float* Wq    = (const float*)d_in[1];
    const float* bq    = (const float*)d_in[2];
    const float* Wk    = (const float*)d_in[3];
    const float* bk    = (const float*)d_in[4];
    const float* Wv    = (const float*)d_in[5];
    const float* bv    = (const float*)d_in[6];
    const float* gamma = (const float*)d_in[7];
    float* out = (float*)d_out;

    // workspace layout (bytes), all 16B-aligned
    char* w = (char*)d_ws;
    __bf16* xt  = (__bf16*)(w);                                   // 8 MB  [b][n][c]
    __bf16* Wqb = (__bf16*)(w + 8388608);                         // 16 KB
    __bf16* Wkb = (__bf16*)(w + 8388608 + 16384);                 // 16 KB
    __bf16* Wvb = (__bf16*)(w + 8388608 + 32768);                 // 128 KB
    __bf16* Qt  = (__bf16*)(w + 8388608 + 163840);                // 1 MB  [b][n][32]
    __bf16* Kt  = (__bf16*)(w + 8388608 + 163840 + 1048576);      // 1 MB  [b][n][32]
    __bf16* Vbf = (__bf16*)(w + 8388608 + 163840 + 2097152);      // 8 MB  [b][c][n]

    // 1) weights -> bf16
    cvt_weights<<<256, 256, 0, stream>>>(Wq, Wk, Wv, Wqb, Wkb, Wvb);

    // 2) x -> xt (transposed bf16)
    transpose_cvt<<<B_DIM * (C_DIM / 64) * (N_PIX / 64), 256, 0, stream>>>(x, xt);

    // 3) projections via WMMA NT-GEMM
    // Qt[n][r] = xt[n][:] . Wq[r][:] + bq[r]   (bias per column)
    gemm_nt_bias<<<dim3(64, B_DIM), 256, 0, stream>>>(
        xt, Wqb, bq, Qt, N_PIX, CR_DIM, C_DIM,
        (long)N_PIX * C_DIM, 0L, (long)N_PIX * CR_DIM, 0);
    gemm_nt_bias<<<dim3(64, B_DIM), 256, 0, stream>>>(
        xt, Wkb, bk, Kt, N_PIX, CR_DIM, C_DIM,
        (long)N_PIX * C_DIM, 0L, (long)N_PIX * CR_DIM, 0);
    // Vb[c][n] = Wv[c][:] . xt[n][:] + bv[c]   (bias per row)
    gemm_nt_bias<<<dim3(512, B_DIM), 256, 0, stream>>>(
        Wvb, xt, bv, Vbf, C_DIM, N_PIX, C_DIM,
        0L, (long)N_PIX * C_DIM, (long)C_DIM * N_PIX, 1);

    // 4) fused relu-attention + L1 norm + residual
    attn_fused<<<dim3(N_PIX / MT, B_DIM), 256, 0, stream>>>(
        Qt, Kt, Vbf, x, gamma, out);
}
